// EncTimeModulatorHandler_39728447488733
// MI455X (gfx1250) — matrix operand order repaired and checked by reading, not compiled, not run on hardware
//
#include <hip/hip_runtime.h>
#include <hip/hip_bf16.h>
#include <cstdint>
#include <cstddef>

typedef __attribute__((ext_vector_type(16))) _Float16 v16h;
typedef __attribute__((ext_vector_type(8)))  _Float16 v8h;
typedef __attribute__((ext_vector_type(8)))  float    v8f;

#define B_   32
#define L_   1024
#define C_   2
#define M_   96
#define E_   1024
#define K_   192      // 2*M (sin || cos features)
#define E2_  2048     // alpha || beta columns
#define CL_  2048     // C*L rows per batch after channel concat
#define TWO_PI_OVER_T 0.006283185307179586f   // 2*pi / T_MAX (T_MAX = 1000)

// ---------------------------------------------------------------------------
// Kernel 1: repack the four fp32 weight tensors into one GEMM-ready f16 tensor
//   Bpack[c][k][e2]:  k<96 -> sin weights (freq m=k), k>=96 -> cos weights
//                     e2<1024 -> alpha, e2>=1024 -> beta
// ---------------------------------------------------------------------------
__global__ void pack_weights(const float* __restrict__ a_sin,
                             const float* __restrict__ a_cos,
                             const float* __restrict__ b_sin,
                             const float* __restrict__ b_cos,
                             _Float16* __restrict__ Bpack) {
    int idx = blockIdx.x * blockDim.x + threadIdx.x;   // < C_*K_*E2_
    int c  = idx / (K_ * E2_);
    int r  = idx - c * (K_ * E2_);
    int k  = r / E2_;
    int e2 = r - k * E2_;
    int m  = (k < M_) ? k : (k - M_);
    int e  = e2 & (E_ - 1);
    const float* src;
    if (e2 < E_) src = (k < M_) ? a_sin : a_cos;
    else         src = (k < M_) ? b_sin : b_cos;
    float v = src[((size_t)c * M_ + m) * E_ + e];
    Bpack[idx] = (_Float16)v;
}

// ---------------------------------------------------------------------------
// Kernel 2: per-batch bitonic argsort of 2048 time keys (index tie-break ->
// total order, deterministic). Emits sorted t, sorted mask, and the inverse
// permutation rank[b][srcRow] = dstRow used by the fused scatter epilogue.
// ---------------------------------------------------------------------------
__global__ __launch_bounds__(1024) void sort_kernel(
    const float* __restrict__ t, const float* __restrict__ mask,
    float* __restrict__ out_t, float* __restrict__ out_m,
    int* __restrict__ rank) {
    __shared__ float skey[CL_];
    __shared__ int   sidx[CL_];
    const int b   = blockIdx.x;
    const int tid = threadIdx.x;

    for (int j = tid; j < CL_; j += 1024) {
        int c = j >> 10, l = j & (L_ - 1);
        skey[j] = t[((size_t)b * L_ + l) * C_ + c];
        sidx[j] = j;
    }
    __syncthreads();

    for (unsigned k = 2; k <= CL_; k <<= 1) {
        for (unsigned jj = k >> 1; jj > 0; jj >>= 1) {
            unsigned i = 2u * tid - (tid & (jj - 1u));  // element with bit jj clear
            unsigned p = i + jj;
            bool asc = ((i & k) == 0u);
            float ki = skey[i], kp = skey[p];
            int   ii = sidx[i], ip = sidx[p];
            bool  gt = (ki > kp) || (ki == kp && ii > ip);
            if (gt == asc) {
                skey[i] = kp; skey[p] = ki;
                sidx[i] = ip; sidx[p] = ii;
            }
            __syncthreads();
        }
    }

    for (int pos = tid; pos < CL_; pos += 1024) {
        int src = sidx[pos];
        int c = src >> 10, l = src & (L_ - 1);
        out_t[(size_t)b * CL_ + pos] = skey[pos];
        out_m[(size_t)b * CL_ + pos] = mask[((size_t)b * L_ + l) * C_ + c];
        rank[(size_t)b * CL_ + src] = pos;
    }
}

// ---------------------------------------------------------------------------
// Kernel 3: fused trig-feature GEMM (f16 WMMA, f32 accum) + modulation +
// rank-permuted scatter store.
//   grid = (row chunks of 256, E-slices of 128, channel); block = 8 waves.
//   B fragments (loop-invariant) are hoisted into 96 VGPRs per wave; A tile
//   is double-buffered in LDS -> exactly one barrier per row-tile.
// ---------------------------------------------------------------------------
#define ROWS_PER_CHUNK 256
#define ROW_TILES      16
#define ESLICE         128
#define COL_STRIDE     200                     // padded halves per column/row
#define BS_BYTES       (256 * COL_STRIDE * 2)  // 102400
#define AS_HALVES      (16 * COL_STRIDE)       // 3200 per buffer
#define AS_OFF         BS_BYTES                // double-buffered A tile
#define TS_OFF         (AS_OFF + 2 * AS_HALVES * 2)   // 115200
#define XS_OFF         (TS_OFF + ROWS_PER_CHUNK * 4)  // 116224
#define RK_OFF         (XS_OFF + ROWS_PER_CHUNK * 4)  // 117248
#define SMEM_BYTES     (RK_OFF + ROWS_PER_CHUNK * 4)  // 118272 (< 320KB/WGP)

__global__ __launch_bounds__(256) void modulate_kernel(
    const float* __restrict__ x, const float* __restrict__ t,
    const float* __restrict__ W_emb, const float* __restrict__ b_emb,
    const _Float16* __restrict__ Bpack, const int* __restrict__ rank,
    float* __restrict__ out_x) {
    extern __shared__ __align__(16) char smem[];
    _Float16* Bs  = (_Float16*)smem;                 // [256 cols][COL_STRIDE]
    _Float16* As  = (_Float16*)(smem + AS_OFF);      // [2][16 rows][COL_STRIDE]
    float*    ts  = (float*)(smem + TS_OFF);         // [256] t per chunk row
    float*    xs  = (float*)(smem + XS_OFF);         // [256] x per chunk row
    int*      rks = (int*)(smem + RK_OFF);           // [256] dst row per chunk row

    const int tid   = threadIdx.x;
    const int chunk = blockIdx.x;            // 0..127  (256 rows each)
    const int eBase = blockIdx.y * ESLICE;   // 0,128,...,896
    const int c     = blockIdx.z;            // channel
    const int b     = (chunk * ROWS_PER_CHUNK) >> 10;  // batch (chunk-uniform)

    // ---- stage B slice into LDS, transposing [k][e2] -> [col][k] ----
    {
        const _Float16* bp = Bpack + (size_t)c * K_ * E2_;
        for (int it = 0; it < 96; ++it) {
            int idx  = it * 256 + tid;       // 0..24575 = 192 k * 128 col-pairs
            int k    = idx >> 7;
            int pair = idx & 127;
            int colLoc, e2;
            if (pair < 64) { colLoc = pair * 2;            e2 = eBase + pair * 2; }
            else           { colLoc = 128 + (pair-64) * 2; e2 = E_ + eBase + (pair-64) * 2; }
            const _Float16* g = bp + (size_t)k * E2_ + e2;
            Bs[(size_t)colLoc       * COL_STRIDE + k] = g[0];
            Bs[(size_t)(colLoc + 1) * COL_STRIDE + k] = g[1];
        }
    }
    // ---- stage per-row metadata for the whole 256-row chunk (coalesced) ----
    {
        int gRow = chunk * ROWS_PER_CHUNK + tid;
        int l    = gRow & (L_ - 1);
        ts[tid]  = t[((size_t)b * L_ + l) * C_ + c];
        xs[tid]  = x[((size_t)b * L_ + l) * C_ + c];
        rks[tid] = rank[(size_t)b * CL_ + c * L_ + l];
    }
    __syncthreads();

    const int  wave = tid >> 5;
    const int  lane = tid & 31;
    const int  nLoc = lane & 15;
    const bool lo   = lane < 16;
    const int  col0 = wave * 16;             // local alpha column base; beta at +128
    const int  e    = eBase + col0 + nLoc;   // global E column for this lane
    const float we  = W_emb[e];
    const float be  = b_emb[e];
    float* outB = out_x + (size_t)b * CL_ * E_ + e;

    // ---- hoist loop-invariant B fragments into registers (96 VGPRs) ----
    // B layout used here (contiguous k, not ISA table order): lo lanes K[0..15], hi lanes K[16..31]; col = lane%16
    v16h bAr[6], bBr[6];
    #pragma unroll
    for (int kk = 0; kk < 6; ++kk) {
        int ko = kk * 32 + (lo ? 0 : 16);
        const _Float16* pA = Bs + (size_t)(col0 + nLoc)       * COL_STRIDE + ko;
        const _Float16* pB = Bs + (size_t)(128 + col0 + nLoc) * COL_STRIDE + ko;
        union { v16h v; v8h h[2]; } u;
        u.h[0] = *(const v8h*)pA;      u.h[1] = *(const v8h*)(pA + 8);
        bAr[kk] = u.v;
        u.h[0] = *(const v8h*)pB;      u.h[1] = *(const v8h*)(pB + 8);
        bBr[kk] = u.v;
    }

    for (int rt = 0; rt < ROW_TILES; ++rt) {
        _Float16* Ab = As + (size_t)(rt & 1) * AS_HALVES;   // ping-pong buffer

        // ---- stage trig A tile (16 rows x 192 K) into current buffer ----
        #pragma unroll
        for (int i = 0; i < 12; ++i) {
            int idx = i * 256 + tid;         // < 3072
            int r   = idx / K_;
            int k   = idx - r * K_;
            float tv = ts[rt * 16 + r];
            int   m  = (k < M_) ? k : (k - M_);
            float ph = tv * (float)m * TWO_PI_OVER_T;
            float v  = (k < M_) ? __sinf(ph) : __cosf(ph);
            Ab[(size_t)r * COL_STRIDE + k] = (_Float16)v;
        }
        __syncthreads();   // single barrier per tile (A is double-buffered)

        // ---- 16x16x192: 6 K-steps, only A reloaded from LDS ----
        v8f accA = {}; v8f accB = {};
        #pragma unroll
        for (int kk = 0; kk < 6; ++kk) {
            const int kb = kk * 32;
            // A layout (matches B): lo lanes K[0..15];
            //                      hi lanes K[16..31]; row = lane%16
            union { v16h v; v8h h[2]; } af;
            const _Float16* pa = Ab + (size_t)nLoc * COL_STRIDE;
            af.h[0] = *(const v8h*)(pa + kb + (lo ? 0 : 16));
            af.h[1] = *(const v8h*)(pa + kb + (lo ? 8 : 24));
            accA = __builtin_amdgcn_wmma_f32_16x16x32_f16(
                false, af.v, false, bAr[kk], (short)0, accA, false, false);
            accB = __builtin_amdgcn_wmma_f32_16x16x32_f16(
                false, af.v, false, bBr[kk], (short)0, accB, false, false);
        }

        // ---- epilogue: mod = (x*W+b)*alpha + beta, scatter to sorted row ----
        // f32 C/D layout: VGPR r <-> row (lo ? r : r+8), col = lane%16
        #pragma unroll
        for (int r = 0; r < 8; ++r) {
            int   rowM   = lo ? r : (r + 8);
            int   row256 = rt * 16 + rowM;
            float xv     = xs[row256];
            int   dstRow = rks[row256];
            float alpha  = accA[r];
            float beta   = accB[r];
            float mod    = (xv * we + be) * alpha + beta;
            outB[(size_t)dstRow * E_] = mod;
        }
    }
}

// ---------------------------------------------------------------------------
extern "C" void kernel_launch(void* const* d_in, const int* in_sizes, int n_in,
                              void* d_out, int out_size, void* d_ws, size_t ws_size,
                              hipStream_t stream) {
    (void)in_sizes; (void)n_in; (void)out_size; (void)ws_size;

    const float* x     = (const float*)d_in[0];
    const float* t     = (const float*)d_in[1];
    const float* mask  = (const float*)d_in[2];
    const float* W_emb = (const float*)d_in[3];
    const float* b_emb = (const float*)d_in[4];
    const float* a_sin = (const float*)d_in[5];
    const float* a_cos = (const float*)d_in[6];
    const float* b_sin = (const float*)d_in[7];
    const float* b_cos = (const float*)d_in[8];

    float* out   = (float*)d_out;
    float* out_x = out;                              // [32][2048][1024]
    float* out_t = out + (size_t)B_ * CL_ * E_;      // [32][2048]
    float* out_m = out_t + (size_t)B_ * CL_;         // [32][2048]

    int* rank = (int*)d_ws;                          // 32*2048 ints
    size_t bp_off = (((size_t)B_ * CL_ * sizeof(int)) + 255) & ~(size_t)255;
    _Float16* Bpack = (_Float16*)((char*)d_ws + bp_off);  // C*192*2048 halves

    pack_weights<<<(C_ * K_ * E2_) / 256, 256, 0, stream>>>(
        a_sin, a_cos, b_sin, b_cos, Bpack);

    sort_kernel<<<B_, 1024, 0, stream>>>(t, mask, out_t, out_m, rank);

    dim3 grid((B_ * L_) / ROWS_PER_CHUNK, E_ / ESLICE, C_);  // (128, 8, 2)
    modulate_kernel<<<grid, 256, SMEM_BYTES, stream>>>(
        x, t, W_emb, b_emb, Bpack, rank, out_x);
}
